// GCN_Layer_4_6408091205940
// MI455X (gfx1250) — matrix operand order repaired and checked by reading, not compiled, not run on hardware
//
#include <hip/hip_runtime.h>
#include <hip/hip_bf16.h>

// ---------------------------------------------------------------------------
// GAT network (4x GATConv + BN + Linear) for MI455X / gfx1250.
// GEMMs use v_wmma_f32_16x16x32_f16 with b128 fragment loads:
//   - A loaded as two contiguous 16B vectors per k-step (one 32B run, k = 16*hs + s, same as packed B)
//   - B pre-packed into WMMA fragment order (one aligned 32B load per frag)
//   - 16x64 wave tiles (4 WMMAs per A fragment) when Nc % 64 == 0
// Edge phase: 3 passes (segment-max, segment-sum, scatter) recomputing the
// cheap per-edge attention score each pass instead of materializing it.
// ---------------------------------------------------------------------------

typedef __attribute__((ext_vector_type(16))) _Float16 v16h;
typedef __attribute__((ext_vector_type(8)))  _Float16 v8h;
typedef __attribute__((ext_vector_type(8)))  float    v8f;

#define NNODES  50000
#define NEDGES  800000
#define NFEAT   128
#define NGRAPHS 50
#define HIDW    256        // H*C for every GAT layer
#define EFULL   (NEDGES + NNODES)

#define CDIV(a, b) (((a) + (b) - 1) / (b))
#define TPB 256

// ------------------------- small helpers -----------------------------------

__device__ __forceinline__ void atomicMaxF(float* addr, float v) {
    // sign-split trick: correct total order for mixed-sign floats
    if (v >= 0.0f) atomicMax((int*)addr, __float_as_int(v));
    else           atomicMin((unsigned int*)addr, __float_as_uint(v));
}

__global__ void k_fill_f32(float* p, float v, long n) {
    long t = (long)blockIdx.x * blockDim.x + threadIdx.x;
    if (t < n) p[t] = v;
}

// ------------------------- self loops / edge concat -------------------------

__global__ void k_selfloop_accum(const int* ei, const float* ea,
                                 float* cnt, float* easum, int E) {
    int t = blockIdx.x * blockDim.x + threadIdx.x;
    if (t >= E) return;
    int d = ei[E + t];                       // dst row of edge_index
    atomicAdd(&cnt[d], 1.0f);
    atomicAdd(&easum[d * 3 + 0], ea[t * 3 + 0]);
    atomicAdd(&easum[d * 3 + 1], ea[t * 3 + 1]);
    atomicAdd(&easum[d * 3 + 2], ea[t * 3 + 2]);
}

__global__ void k_build_edges(const int* ei, const float* ea,
                              const float* cnt, const float* easum,
                              int* srcf, int* dstf, float* eaf, int E, int N) {
    int t = blockIdx.x * blockDim.x + threadIdx.x;
    if (t >= E + N) return;
    if (t < E) {
        srcf[t] = ei[t];
        dstf[t] = ei[E + t];
        eaf[t * 3 + 0] = ea[t * 3 + 0];
        eaf[t * 3 + 1] = ea[t * 3 + 1];
        eaf[t * 3 + 2] = ea[t * 3 + 2];
    } else {
        int n = t - E;
        srcf[t] = n;
        dstf[t] = n;
        float inv = 1.0f / fmaxf(cnt[n], 1.0f);
        eaf[t * 3 + 0] = easum[n * 3 + 0] * inv;
        eaf[t * 3 + 1] = easum[n * 3 + 1] * inv;
        eaf[t * 3 + 2] = easum[n * 3 + 2] * inv;
    }
}

// ------------------------- f32 -> f16 staging -------------------------------

__global__ void k_cvt_a(const float* __restrict__ x, _Float16* __restrict__ ah,
                        long M, int K, int Kpad) {
    long t = (long)blockIdx.x * blockDim.x + threadIdx.x;
    long total = M * Kpad;
    if (t >= total) return;
    int k = (int)(t % Kpad);
    long n = t / Kpad;
    ah[t] = (k < K) ? (_Float16)x[n * K + k] : (_Float16)0.0f;
}

// W is (Nout, Kin) row-major. Pack B = W^T (padded to Kpad rows) directly into
// WMMA fragment order: frag(tn16, kblk) = 32 lanes x 16 contiguous halves,
//   b[j] = B[(kblk*32 + (lane>>4)*16 + j)*Nout + tn16*16 + (lane&15)]
__global__ void k_cvt_b_pack(const float* __restrict__ W, _Float16* __restrict__ bp,
                             int Nout, int Kin, int Kpad) {
    int kblks = Kpad >> 5;
    int total = (Nout >> 4) * kblks * 512;          // 32 lanes * 16 halves
    int t = blockIdx.x * blockDim.x + threadIdx.x;
    if (t >= total) return;
    int j    = t & 15;
    int lane = (t >> 4) & 31;
    int rest = t >> 9;
    int kblk = rest % kblks;
    int tn16 = rest / kblks;
    int n = (tn16 << 4) + (lane & 15);
    int k = (kblk << 5) + ((lane >> 4) << 4) + j;
    bp[t] = (k < Kin) ? (_Float16)W[n * Kin + k] : (_Float16)0.0f;
}

// ------------------------- WMMA GEMM ---------------------------------------
// C[M,Nc] = A[M,K] * Bpacked, f16 in / f32 out. One wave per 16x(16*NT) tile.
// M % 16 == 0, Nc % (16*NT) == 0, K % 32 == 0 (caller pads).
template <int NT>
__global__ void k_gemm_wmma(const _Float16* __restrict__ A,
                            const _Float16* __restrict__ Bp,
                            float* __restrict__ C, int M, int Nc, int K) {
    int wave = (int)(((long)blockIdx.x * blockDim.x + threadIdx.x) >> 5);
    int lane = threadIdx.x & 31;
    int tilesN = Nc >> 4;                 // 16-wide tile columns
    int wTilesN = tilesN / NT;            // wave tile columns
    int tm   = (wave / wTilesN) << 4;
    int tn16 = (wave % wTilesN) * NT;     // first 16-col tile index
    if (tm >= M) return;                  // wave-uniform: EXEC stays all-ones

    v8f acc[NT];
#pragma unroll
    for (int t = 0; t < NT; t++) acc[t] = (v8f){0.f, 0.f, 0.f, 0.f, 0.f, 0.f, 0.f, 0.f};

    const int m  = tm + (lane & 15);
    const int hs = lane >> 4;             // half-wave select
    const int kblks = K >> 5;
    const v8h*  arow  = (const v8h*)(A + (long)m * K);   // 8-half units
    const v16h* bfrag = (const v16h*)Bp;                 // 16-half units

    for (int k0 = 0; k0 < K; k0 += 32) {
        // A fragment: one contiguous 32B run, k = 16*hs + s (matches packed B)
        v8h a0 = arow[(k0 >> 3) + 2 * hs];
        v8h a1 = arow[(k0 >> 3) + 2 * hs + 1];
        v16h a = __builtin_shufflevector(a0, a1,
                 0, 1, 2, 3, 4, 5, 6, 7, 8, 9, 10, 11, 12, 13, 14, 15);
#pragma unroll
        for (int t = 0; t < NT; t++) {
            v16h b = bfrag[((long)(tn16 + t) * kblks + (k0 >> 5)) * 32 + lane];
            acc[t] = __builtin_amdgcn_wmma_f32_16x16x32_f16(
                /*neg_a=*/false, a, /*neg_b=*/false, b,
                /*c_mod=*/(short)0, acc[t], /*reuse_a=*/false, /*reuse_b=*/false);
        }
    }
    int row0 = tm + (hs << 3);
#pragma unroll
    for (int t = 0; t < NT; t++) {
        int col = ((tn16 + t) << 4) + (lane & 15);
#pragma unroll
        for (int v = 0; v < 8; v++)
            C[(long)(row0 + v) * Nc + col] = acc[t][v];
    }
}

// ------------------------- attention scalars -------------------------------

// per (node, head): dots of xs row-slice with att_src / att_dst
__global__ void k_att_node(const float* __restrict__ xs,
                           const float* __restrict__ asrc,
                           const float* __restrict__ adst,
                           float* __restrict__ s_, float* __restrict__ d_,
                           int N, int H, int C) {
    int t = blockIdx.x * blockDim.x + threadIdx.x;
    if (t >= N * H) return;
    int h = t % H;
    long n = t / H;
    const float* row = xs + n * HIDW + h * C;
    float ss = 0.f, sd = 0.f;
    for (int c = 0; c < C; c++) {
        float v = row[c];
        ss += v * asrc[h * C + c];
        sd += v * adst[h * C + c];
    }
    s_[n * H + h] = ss;
    d_[n * H + h] = sd;
}

// qe[h,j] = sum_c we[(h*C+c)*3 + j] * att_edge[h*C+c]   (folds lin_edge)
__global__ void k_qe(const float* __restrict__ we, const float* __restrict__ ae,
                     float* __restrict__ qe, int H, int C) {
    int t = threadIdx.x;
    if (t >= H * 3) return;
    int j = t % 3, h = t / 3;
    float s = 0.f;
    for (int c = 0; c < C; c++)
        s += we[(h * C + c) * 3 + j] * ae[h * C + c];
    qe[h * 3 + j] = s;
}

__device__ __forceinline__ float edge_score(long e, int h,
                                            const int* srcf, const int* dstf,
                                            const float* eaf,
                                            const float* s_, const float* d_,
                                            const float* qe, int H) {
    int s = srcf[e], d = dstf[e];
    float a = s_[(long)s * H + h] + d_[(long)d * H + h]
            + eaf[e * 3 + 0] * qe[h * 3 + 0]
            + eaf[e * 3 + 1] * qe[h * 3 + 1]
            + eaf[e * 3 + 2] * qe[h * 3 + 2];
    return (a > 0.f) ? a : 0.2f * a;     // LeakyReLU(0.2)
}

__global__ void k_edge_max(const int* srcf, const int* dstf, const float* eaf,
                           const float* s_, const float* d_, const float* qe,
                           float* mx, int EF, int H) {
    long t = (long)blockIdx.x * blockDim.x + threadIdx.x;
    if (t >= (long)EF * H) return;
    int h = (int)(t % H);
    long e = t / H;
    float a = edge_score(e, h, srcf, dstf, eaf, s_, d_, qe, H);
    atomicMaxF(&mx[(long)dstf[e] * H + h], a);
}

__global__ void k_edge_den(const int* srcf, const int* dstf, const float* eaf,
                           const float* s_, const float* d_, const float* qe,
                           const float* mx, float* den, int EF, int H) {
    long t = (long)blockIdx.x * blockDim.x + threadIdx.x;
    if (t >= (long)EF * H) return;
    int h = (int)(t % H);
    long e = t / H;
    float a = edge_score(e, h, srcf, dstf, eaf, s_, d_, qe, H);
    float ex = __expf(a - mx[(long)dstf[e] * H + h]);
    atomicAdd(&den[(long)dstf[e] * H + h], ex);
}

// one wave per edge; each lane covers 8 of the 256 hc channels
__global__ void k_edge_scatter(const int* srcf, const int* dstf, const float* eaf,
                               const float* s_, const float* d_, const float* qe,
                               const float* mx, const float* den,
                               const float* __restrict__ xs, float* __restrict__ out,
                               int EF, int H, int logC) {
    long gtid = (long)blockIdx.x * blockDim.x + threadIdx.x;
    long e = gtid >> 5;
    int lane = (int)(gtid & 31);
    if (e >= EF) return;                 // wave-uniform
    int s = srcf[e], d = dstf[e];
#pragma unroll
    for (int j = 0; j < 8; j++) {
        int hc = lane + (j << 5);
        int h = hc >> logC;
        float a = edge_score(e, h, srcf, dstf, eaf, s_, d_, qe, H);
        float ex = __expf(a - mx[(long)d * H + h]);
        float alpha = ex / (den[(long)d * H + h] + 1e-16f);
        atomicAdd(&out[(long)d * HIDW + hc], alpha * xs[(long)s * HIDW + hc]);
    }
}

// ------------------------- bias / BN / pool / head --------------------------

__global__ void k_bias(float* h, const float* b, long total, int W) {
    long t = (long)blockIdx.x * blockDim.x + threadIdx.x;
    if (t < total) h[t] += b[t % W];
}

__global__ void k_bn_stats(const float* __restrict__ x, float* __restrict__ stat,
                           int N, int W) {
    int col = blockIdx.x;                // one block per column
    float s = 0.f, s2 = 0.f;
    for (int r = threadIdx.x; r < N; r += blockDim.x) {
        float v = x[(long)r * W + col];
        s += v; s2 += v * v;
    }
    __shared__ float sh[TPB], sh2[TPB];
    sh[threadIdx.x] = s; sh2[threadIdx.x] = s2;
    __syncthreads();
    for (int o = blockDim.x >> 1; o > 0; o >>= 1) {
        if (threadIdx.x < o) { sh[threadIdx.x] += sh[threadIdx.x + o];
                               sh2[threadIdx.x] += sh2[threadIdx.x + o]; }
        __syncthreads();
    }
    if (threadIdx.x == 0) { stat[col] = sh[0]; stat[W + col] = sh2[0]; }
}

__global__ void k_bn_apply(float* __restrict__ h, const float* __restrict__ stat,
                           const float* __restrict__ g, const float* __restrict__ b,
                           long N, int W, int relu) {
    long t = (long)blockIdx.x * blockDim.x + threadIdx.x;
    if (t >= N * W) return;
    int col = (int)(t % W);
    float mu = stat[col] / (float)N;
    float var = stat[W + col] / (float)N - mu * mu;
    float y = (h[t] - mu) * rsqrtf(var + 1e-5f) * g[col] + b[col];
    if (relu) y = fmaxf(y, 0.f);
    h[t] = y;
}

__global__ void k_pool(const float* __restrict__ h, const int* __restrict__ batch,
                       float* __restrict__ pooled, float* __restrict__ gcnt, int N) {
    long t = (long)blockIdx.x * blockDim.x + threadIdx.x;
    if (t >= (long)N * HIDW) return;
    int w = (int)(t % HIDW);
    long n = t / HIDW;
    int g = batch[n];
    atomicAdd(&pooled[(long)g * HIDW + w], h[t]);
    if (w == 0) atomicAdd(&gcnt[g], 1.0f);
}

__global__ void k_head(const float* __restrict__ pooled, const float* __restrict__ gcnt,
                       const float* __restrict__ Wo, const float* __restrict__ bo,
                       float* __restrict__ out) {
    int t = blockIdx.x * blockDim.x + threadIdx.x;
    if (t >= NGRAPHS * 10) return;
    int c = t % 10, g = t / 10;
    float inv = 1.0f / fmaxf(gcnt[g], 1.0f);
    float s = bo[c];
    for (int k = 0; k < HIDW; k++)
        s += pooled[g * HIDW + k] * inv * Wo[c * HIDW + k];
    out[t] = s;
}

// ------------------------- host orchestration -------------------------------

struct Ws {
    float* xs; float* gout; _Float16* ah; _Float16* bh;
    float* as_; float* ad_; float* mx; float* den;
    int* srcf; int* dstf; float* eaf;
    float* cnt; float* easum; float* qe; float* stat;
    float* pooled; float* gcnt;
};

static char* wspush(char*& p, size_t bytes) {
    char* r = p;
    p += (bytes + 255) & ~(size_t)255;
    return r;
}

static void run_gemm(const float* x, const float* W, float* C,
                     int M, int Nout, int Kin, Ws& w, hipStream_t st) {
    int Kpad = (Kin + 31) & ~31;
    long atot = (long)M * Kpad;
    k_cvt_a<<<(int)CDIV(atot, TPB), TPB, 0, st>>>(x, w.ah, M, Kin, Kpad);
    int btot = (Nout >> 4) * (Kpad >> 5) * 512;
    k_cvt_b_pack<<<CDIV(btot, TPB), TPB, 0, st>>>(W, w.bh, Nout, Kin, Kpad);
    if ((Nout & 63) == 0) {
        int tiles = (M / 16) * (Nout / 64);
        k_gemm_wmma<4><<<CDIV(tiles, 8), TPB, 0, st>>>(w.ah, w.bh, C, M, Nout, Kpad);
    } else {
        int tiles = (M / 16) * (Nout / 16);
        k_gemm_wmma<1><<<CDIV(tiles, 8), TPB, 0, st>>>(w.ah, w.bh, C, M, Nout, Kpad);
    }
}

static void run_bn(float* h, const float* g, const float* b, int W, int relu,
                   Ws& w, hipStream_t st) {
    k_bn_stats<<<W, TPB, 0, st>>>(h, w.stat, NNODES, W);
    long tot = (long)NNODES * W;
    k_bn_apply<<<(int)CDIV(tot, TPB), TPB, 0, st>>>(h, w.stat, g, b, NNODES, W, relu);
}

// one GATConv: xin [N, fin] -> gout [N, 256]
static void run_gat(const float* xin, const float* const* P, int fin, int H, int C,
                    Ws& w, hipStream_t st) {
    const float *Wm = P[0], *We = P[1], *Asrc = P[2], *Adst = P[3], *Ae = P[4], *Bias = P[5];
    // xs = xin @ W^T
    run_gemm(xin, Wm, w.xs, NNODES, HIDW, fin, w, st);
    // per-node / per-head attention scalars
    k_att_node<<<CDIV(NNODES * H, TPB), TPB, 0, st>>>(w.xs, Asrc, Adst, w.as_, w.ad_, NNODES, H, C);
    k_qe<<<1, 64, 0, st>>>(We, Ae, w.qe, H, C);
    // segment softmax over dst + scatter
    k_fill_f32<<<CDIV(NNODES * H, TPB), TPB, 0, st>>>(w.mx, -INFINITY, (long)NNODES * H);
    k_fill_f32<<<CDIV(NNODES * H, TPB), TPB, 0, st>>>(w.den, 0.f, (long)NNODES * H);
    k_fill_f32<<<(int)CDIV((long)NNODES * HIDW, TPB), TPB, 0, st>>>(w.gout, 0.f, (long)NNODES * HIDW);
    long eh = (long)EFULL * H;
    k_edge_max<<<(int)CDIV(eh, TPB), TPB, 0, st>>>(w.srcf, w.dstf, w.eaf, w.as_, w.ad_, w.qe, w.mx, EFULL, H);
    k_edge_den<<<(int)CDIV(eh, TPB), TPB, 0, st>>>(w.srcf, w.dstf, w.eaf, w.as_, w.ad_, w.qe, w.mx, w.den, EFULL, H);
    int logC = (C == 16) ? 4 : (C == 32) ? 5 : (C == 64) ? 6 : 7;
    long sthreads = (long)EFULL * 32;
    k_edge_scatter<<<(int)CDIV(sthreads, TPB), TPB, 0, st>>>(w.srcf, w.dstf, w.eaf, w.as_, w.ad_,
                                                             w.qe, w.mx, w.den, w.xs, w.gout,
                                                             EFULL, H, logC);
    k_bias<<<(int)CDIV((long)NNODES * HIDW, TPB), TPB, 0, st>>>(w.gout, Bias, (long)NNODES * HIDW, HIDW);
}

extern "C" void kernel_launch(void* const* d_in, const int* in_sizes, int n_in,
                              void* d_out, int out_size, void* d_ws, size_t ws_size,
                              hipStream_t stream) {
    // ---- input mapping (setup_inputs dict insertion order, recursive) ----
    const float* x      = (const float*)d_in[0];   // [50000,128]
    const int*   ei     = (const int*)  d_in[1];   // [2,800000]
    const float* ea     = (const float*)d_in[2];   // [800000,3]
    const int*   batch  = (const int*)  d_in[3];   // [50000]
    // params: gat{w,we,att_src,att_dst,att_edge,bias}, bn{g,b}, lin{w,b}, bnl{g,b}, ...
    const float* gat1[6]; const float* gat2[6]; const float* gat3[6]; const float* gat4[6];
    int i = 4;
    for (int j = 0; j < 6; j++) gat1[j] = (const float*)d_in[i++];
    const float *bn1g = (const float*)d_in[i++], *bn1b = (const float*)d_in[i++];
    const float *l1w  = (const float*)d_in[i++], *l1b  = (const float*)d_in[i++];
    const float *bl1g = (const float*)d_in[i++], *bl1b = (const float*)d_in[i++];
    for (int j = 0; j < 6; j++) gat2[j] = (const float*)d_in[i++];
    const float *bn2g = (const float*)d_in[i++], *bn2b = (const float*)d_in[i++];
    const float *l2w  = (const float*)d_in[i++], *l2b  = (const float*)d_in[i++];
    const float *bl2g = (const float*)d_in[i++], *bl2b = (const float*)d_in[i++];
    for (int j = 0; j < 6; j++) gat3[j] = (const float*)d_in[i++];
    const float *bn3g = (const float*)d_in[i++], *bn3b = (const float*)d_in[i++];
    const float *l3w  = (const float*)d_in[i++], *l3b  = (const float*)d_in[i++];
    const float *bl3g = (const float*)d_in[i++], *bl3b = (const float*)d_in[i++];
    for (int j = 0; j < 6; j++) gat4[j] = (const float*)d_in[i++];
    const float *bn4g = (const float*)d_in[i++], *bn4b = (const float*)d_in[i++];
    const float *ow   = (const float*)d_in[i++], *ob   = (const float*)d_in[i++];

    // ---- workspace carve ----
    char* p = (char*)d_ws;
    Ws w;
    w.xs     = (float*)    wspush(p, (size_t)NNODES * HIDW * 4);
    w.gout   = (float*)    wspush(p, (size_t)NNODES * HIDW * 4);
    w.ah     = (_Float16*) wspush(p, (size_t)NNODES * HIDW * 2);
    w.bh     = (_Float16*) wspush(p, (size_t)256 * 256 * 2);
    w.as_    = (float*)    wspush(p, (size_t)NNODES * 16 * 4);
    w.ad_    = (float*)    wspush(p, (size_t)NNODES * 16 * 4);
    w.mx     = (float*)    wspush(p, (size_t)NNODES * 16 * 4);
    w.den    = (float*)    wspush(p, (size_t)NNODES * 16 * 4);
    w.srcf   = (int*)      wspush(p, (size_t)EFULL * 4);
    w.dstf   = (int*)      wspush(p, (size_t)EFULL * 4);
    w.eaf    = (float*)    wspush(p, (size_t)EFULL * 3 * 4);
    w.cnt    = (float*)    wspush(p, (size_t)NNODES * 4);
    w.easum  = (float*)    wspush(p, (size_t)NNODES * 3 * 4);
    w.qe     = (float*)    wspush(p, 64 * 4);
    w.stat   = (float*)    wspush(p, 512 * 4);
    w.pooled = (float*)    wspush(p, (size_t)NGRAPHS * HIDW * 4);
    w.gcnt   = (float*)    wspush(p, (size_t)NGRAPHS * 4);

    hipStream_t st = stream;

    // ---- self-loop edge_attr (per-dst mean) + concatenated edge list ----
    k_fill_f32<<<CDIV(NNODES, TPB), TPB, 0, st>>>(w.cnt, 0.f, NNODES);
    k_fill_f32<<<CDIV(NNODES * 3, TPB), TPB, 0, st>>>(w.easum, 0.f, (long)NNODES * 3);
    k_selfloop_accum<<<CDIV(NEDGES, TPB), TPB, 0, st>>>(ei, ea, w.cnt, w.easum, NEDGES);
    k_build_edges<<<CDIV(EFULL, TPB), TPB, 0, st>>>(ei, ea, w.cnt, w.easum,
                                                    w.srcf, w.dstf, w.eaf, NEDGES, NNODES);

    // ---- layer 1: GAT(128 -> 16x16) ----
    run_gat(x, gat1, NFEAT, 16, 16, w, st);
    run_bn(w.gout, bn1g, bn1b, HIDW, 1, w, st);
    run_gemm(w.gout, l1w, w.xs, NNODES, 16, HIDW, w, st);         // lin1 -> xs
    k_bias<<<(int)CDIV((long)NNODES * 16, TPB), TPB, 0, st>>>(w.xs, l1b, (long)NNODES * 16, 16);
    run_bn(w.xs, bl1g, bl1b, 16, 1, w, st);

    // ---- layer 2: GAT(16 -> 8x32) ----
    run_gat(w.xs, gat2, 16, 8, 32, w, st);
    run_bn(w.gout, bn2g, bn2b, HIDW, 1, w, st);
    run_gemm(w.gout, l2w, w.xs, NNODES, 32, HIDW, w, st);         // lin2 -> xs
    k_bias<<<(int)CDIV((long)NNODES * 32, TPB), TPB, 0, st>>>(w.xs, l2b, (long)NNODES * 32, 32);
    run_bn(w.xs, bl2g, bl2b, 32, 1, w, st);

    // ---- layer 3: GAT(32 -> 4x64) ----
    run_gat(w.xs, gat3, 32, 4, 64, w, st);
    run_bn(w.gout, bn3g, bn3b, HIDW, 1, w, st);
    run_gemm(w.gout, l3w, w.xs, NNODES, 64, HIDW, w, st);         // lin3 -> xs
    k_bias<<<(int)CDIV((long)NNODES * 64, TPB), TPB, 0, st>>>(w.xs, l3b, (long)NNODES * 64, 64);
    run_bn(w.xs, bl3g, bl3b, 64, 1, w, st);

    // ---- layer 4: GAT(64 -> 2x128) ----
    run_gat(w.xs, gat4, 64, 2, 128, w, st);
    run_bn(w.gout, bn4g, bn4b, HIDW, 1, w, st);

    // ---- mean pool per graph + classifier head ----
    k_fill_f32<<<CDIV(NGRAPHS * HIDW, TPB), TPB, 0, st>>>(w.pooled, 0.f, (long)NGRAPHS * HIDW);
    k_fill_f32<<<1, TPB, 0, st>>>(w.gcnt, 0.f, NGRAPHS);
    k_pool<<<(int)CDIV((long)NNODES * HIDW, TPB), TPB, 0, st>>>(w.gout, batch, w.pooled, w.gcnt, NNODES);
    k_head<<<CDIV(NGRAPHS * 10, TPB), TPB, 0, st>>>(w.pooled, w.gcnt, ow, ob, (float*)d_out);
}
